// MLPFusionHiRA_24206435680546
// MI455X (gfx1250) — matrix operand order repaired and checked by reading, not compiled, not run on hardware
//
#include <hip/hip_runtime.h>

typedef unsigned short u16;
typedef __attribute__((ext_vector_type(16))) __bf16 v16bf;
typedef __attribute__((ext_vector_type(8)))  __bf16 bf16x8;
typedef __attribute__((ext_vector_type(8)))  float  v8f;

#define M_MOD 4
#define B_DIM 4
#define N_TOK 1024
#define C_DIM 768
#define H_DIM 3072
#define R_DIM 16
#define ROWS (M_MOD * B_DIM * N_TOK)       /* 16384 */
#define ROWS_PER_MOD (B_DIM * N_TOK)       /* 4096  */

__device__ __forceinline__ u16 f2bf(float f) {
  unsigned u = __float_as_uint(f);
  u += 0x7FFFu + ((u >> 16) & 1u);          // round-to-nearest-even
  return (u16)(u >> 16);
}
__device__ __forceinline__ float bf2f(u16 h) {
  return __uint_as_float(((unsigned)h) << 16);
}
__device__ __forceinline__ float gelu_exact(float x) {
  return 0.5f * x * (1.0f + erff(x * 0.70710678118654752f));
}

// CDNA5 async memory->LDS copy (ASYNCcnt-tracked), GV addressing mode.
__device__ __forceinline__ void async_g2l_b128(const u16* lds_dst, const u16* gsrc) {
  unsigned loff =
      (unsigned)(unsigned long long)(__attribute__((address_space(3))) const u16*)lds_dst;
  asm volatile("global_load_async_to_lds_b128 %0, %1, off"
               :: "v"(loff), "v"(gsrc)
               : "memory");
}
#define WAIT_ASYNC(n) asm volatile("s_wait_asynccnt " #n ::: "memory")

// ---------------------------------------------------------------- prep kernels

__global__ void cast_f32_to_bf16_kernel(const float* __restrict__ src,
                                        u16* __restrict__ dst, int n4) {
  int t = blockIdx.x * blockDim.x + threadIdx.x;
  if (t >= n4) return;
  float4 v = ((const float4*)src)[t];
  u16 o[4] = { f2bf(v.x), f2bf(v.y), f2bf(v.z), f2bf(v.w) };
  ((ulonglong1*)dst)[t] = *(ulonglong1*)o;  // 8-byte store
}

// delta[k,row,col] = W[row,col] * sum_r Bm[k,row,r] * Am[k,r,col]   (bf16 out)
__global__ void make_delta_kernel(const float* __restrict__ Wf,
                                  const float* __restrict__ Am,  // [4, R_DIM, Ccols]
                                  const float* __restrict__ Bm,  // [4, Rrows, R_DIM]
                                  u16* __restrict__ dout, int Rrows, int Ccols) {
  size_t t = (size_t)blockIdx.x * blockDim.x + threadIdx.x;
  size_t total = (size_t)M_MOD * Rrows * Ccols;
  if (t >= total) return;
  int c = (int)(t % Ccols);
  size_t q = t / Ccols;
  int row = (int)(q % Rrows);
  int k = (int)(q / Rrows);
  const float* bp = Bm + ((size_t)k * Rrows + row) * R_DIM;
  const float* ap = Am + (size_t)k * R_DIM * Ccols + c;
  float s = 0.f;
#pragma unroll
  for (int r = 0; r < R_DIM; ++r) s += bp[r] * ap[(size_t)r * Ccols];
  dout[t] = f2bf(Wf[(size_t)row * Ccols + c] * s);
}

// logits[tok*4+e] = dot(x[tok,:], gate_w[i,e,:]) + gate_b[i,e]   (fp32, like ref)
__global__ void gate_logits_kernel(const float* __restrict__ x,
                                   const float* __restrict__ gw,
                                   const float* __restrict__ gb,
                                   float* __restrict__ logits) {
  int t = blockIdx.x * blockDim.x + threadIdx.x;   // ROWS*4
  if (t >= ROWS * M_MOD) return;
  int e = t & 3;
  int tok = t >> 2;
  int i = tok >> 12;                               // tok / 4096
  const float* xr = x + (size_t)tok * C_DIM;
  const float* wr = gw + ((size_t)i * M_MOD + e) * C_DIM;
  float s = gb[i * M_MOD + e];
  for (int c = 0; c < C_DIM; ++c) s += xr[c] * wr[c];
  logits[t] = s;
}

// softmax over experts, expert-mask + renorm, times own-modality mask
__global__ void gate_norm_kernel(const float* __restrict__ logits,
                                 const int* __restrict__ mask,   // [M,B]
                                 float* __restrict__ wn) {
  int tok = blockIdx.x * blockDim.x + threadIdx.x;
  if (tok >= ROWS) return;
  int i = tok >> 12;
  int b = (tok >> 10) & 3;
  float l[4], mk[4];
#pragma unroll
  for (int k = 0; k < 4; ++k) { l[k] = logits[tok * 4 + k]; mk[k] = (float)mask[k * B_DIM + b]; }
  float mx = fmaxf(fmaxf(l[0], l[1]), fmaxf(l[2], l[3]));
  float s = 0.f;
#pragma unroll
  for (int k = 0; k < 4; ++k) { l[k] = __expf(l[k] - mx); s += l[k]; }
  float inv = 1.0f / s;
  float msum = 0.f;
#pragma unroll
  for (int k = 0; k < 4; ++k) { l[k] = l[k] * inv * mk[k]; msum += l[k]; }
  float rn = 1.0f / (msum + 1e-6f);
  float own = (float)mask[i * B_DIM + b];
#pragma unroll
  for (int k = 0; k < 4; ++k) wn[tok * 4 + k] = l[k] * rn * own;
}

// ---------------------------------------------------------------- WMMA GEMM

#define LSTR 40   // LDS row stride in bf16 elems (32 + 8 pad, keeps 16B align)

__device__ __forceinline__ v16bf load_frag_a(const u16* lds, int m0) {
  // 16x32 bf16 A frag: lane = row (mod 16); K-half by lane>>4 (ISA 7.12.2)
  int lane = threadIdx.x & 31;
  int r = m0 + (lane & 15);
  int kh = (lane >> 4) * 8;
  union { v16bf v; bf16x8 h[2]; } u;
  u.h[0] = *(const bf16x8*)(lds + r * LSTR + kh);
  u.h[1] = *(const bf16x8*)(lds + r * LSTR + kh + 16);
  return u.v;
}
__device__ __forceinline__ v16bf load_frag_b(const u16* lds, int n0) {
  // 32x16 bf16 B frag from row-major W tile: lane = column (= W row), K-half by lane>>4 (same order as A)
  int lane = threadIdx.x & 31;
  int r = n0 + (lane & 15);
  int kh = (lane >> 4) * 8;
  union { v16bf v; bf16x8 h[2]; } u;
  u.h[0] = *(const bf16x8*)(lds + r * LSTR + kh);
  u.h[1] = *(const bf16x8*)(lds + r * LSTR + kh + 16);
  return u.v;
}

// C[rows,N] = A[rows,K] @ W^T; dual output: W0 (shared) and Wd[mod] (per-modality).
// BM=128 x BN=128 block tile, 8 waves in 2x4 grid, wave tile 64x32, BK=32,
// double-buffered LDS filled by global_load_async_to_lds_b128 (ASYNCcnt pipeline).
template <bool OUT_BF16>
__global__ __launch_bounds__(256) void gemm_dual_kernel(
    const u16* __restrict__ Abf, const u16* __restrict__ W0,
    const u16* __restrict__ Wd, void* __restrict__ out0, void* __restrict__ out1,
    int K, int N) {
  constexpr int BM = 128, BN = 128, BK = 32;
  __shared__ u16 sA[2][BM * LSTR];
  __shared__ u16 sB0[2][BN * LSTR];
  __shared__ u16 sB1[2][BN * LSTR];

  const int tid = threadIdx.x;
  const int wave = tid >> 5;                 // 0..7  (wave32)
  const int row0 = blockIdx.x * BM;          // x fastest => blocks sharing weights adjacent
  const int col0 = blockIdx.y * BN;
  const int mod = row0 / ROWS_PER_MOD;       // 128 | 4096 => uniform per block
  const u16* W1 = Wd + (size_t)mod * N * K;

  const int wm = (wave >> 2) * 64;           // 2 wave-rows x 4 wave-cols, 64x32 each
  const int wn = (wave & 3) * 32;

  v8f acc0[4][2], acc1[4][2];
#pragma unroll
  for (int a = 0; a < 4; ++a)
#pragma unroll
    for (int b = 0; b < 2; ++b) {
      acc0[a][b] = v8f{0.f, 0.f, 0.f, 0.f, 0.f, 0.f, 0.f, 0.f};
      acc1[a][b] = v8f{0.f, 0.f, 0.f, 0.f, 0.f, 0.f, 0.f, 0.f};
    }

  // staging: 128 rows x 32 cols = 512 x 16B chunks per tile, 2 per thread
  auto issue_tile = [&](int buf, int k0) {
#pragma unroll
    for (int p = 0; p < 2; ++p) {
      int idx = tid + p * 256;
      int r = idx >> 2;
      int ck = (idx & 3) * 8;
      async_g2l_b128(&sA[buf][r * LSTR + ck],
                     Abf + (size_t)(row0 + r) * K + k0 + ck);
      size_t g = (size_t)(col0 + r) * K + k0 + ck;
      async_g2l_b128(&sB0[buf][r * LSTR + ck], W0 + g);
      async_g2l_b128(&sB1[buf][r * LSTR + ck], W1 + g);
    }
  };

  const int nk = K / BK;
  issue_tile(0, 0);
  for (int it = 0; it < nk; ++it) {
    const int buf = it & 1;
    const bool more = (it + 1 < nk);
    if (more) issue_tile(buf ^ 1, (it + 1) * BK);
    if (more) { WAIT_ASYNC(0x6); } else { WAIT_ASYNC(0x0); }
    __syncthreads();                         // tile `buf` visible to all waves

    v16bf af[4];
#pragma unroll
    for (int mi = 0; mi < 4; ++mi) af[mi] = load_frag_a(&sA[buf][0], wm + mi * 16);
#pragma unroll
    for (int ni = 0; ni < 2; ++ni) {
      v16bf b0 = load_frag_b(&sB0[buf][0], wn + ni * 16);
      v16bf b1 = load_frag_b(&sB1[buf][0], wn + ni * 16);
#pragma unroll
      for (int mi = 0; mi < 4; ++mi) {
        acc0[mi][ni] = __builtin_amdgcn_wmma_f32_16x16x32_bf16(
            false, af[mi], false, b0, (short)0, acc0[mi][ni], false, false);
        acc1[mi][ni] = __builtin_amdgcn_wmma_f32_16x16x32_bf16(
            false, af[mi], false, b1, (short)0, acc1[mi][ni], false, false);
      }
    }
    __syncthreads();                         // done reading `buf` before refill
  }

  // epilogue: C/D layout (ISA 7.12.2): row = v + 8*(lane>>4), col = lane&15
  const int lane = tid & 31;
  const int cl = lane & 15;
  const int rh = (lane >> 4) * 8;
#pragma unroll
  for (int mi = 0; mi < 4; ++mi)
#pragma unroll
    for (int ni = 0; ni < 2; ++ni) {
      int gr = row0 + wm + mi * 16 + rh;
      int gc = col0 + wn + ni * 16 + cl;
#pragma unroll
      for (int v = 0; v < 8; ++v) {
        size_t o = (size_t)(gr + v) * N + gc;
        if constexpr (OUT_BF16) {
          ((u16*)out0)[o] = f2bf(acc0[mi][ni][v]);
          ((u16*)out1)[o] = f2bf(acc1[mi][ni][v]);
        } else {
          ((float*)out0)[o] = acc0[mi][ni][v];
          ((float*)out1)[o] = acc1[mi][ni][v];
        }
      }
    }
}

// ---------------------------------------------------------------- combines

__global__ void combine1_kernel(const u16* __restrict__ base,
                                const u16* __restrict__ lora,
                                const float* __restrict__ wn,
                                const float* __restrict__ bias,
                                u16* __restrict__ x1) {
  size_t t = (size_t)blockIdx.x * blockDim.x + threadIdx.x;
  const int chunks = H_DIM / 8;  // 384
  if (t >= (size_t)ROWS * chunks) return;
  int r = (int)(t / chunks);
  int ch = (int)(t % chunks) * 8;
  int bn = r & (ROWS_PER_MOD - 1);
  float w0 = wn[r * 4 + 0], w1 = wn[r * 4 + 1], w2 = wn[r * 4 + 2], w3 = wn[r * 4 + 3];
  const u16* bp = base + (size_t)r * H_DIM + ch;
  const u16* l0 = lora + ((size_t)(0 * ROWS_PER_MOD + bn)) * H_DIM + ch;
  const u16* l1 = lora + ((size_t)(1 * ROWS_PER_MOD + bn)) * H_DIM + ch;
  const u16* l2 = lora + ((size_t)(2 * ROWS_PER_MOD + bn)) * H_DIM + ch;
  const u16* l3 = lora + ((size_t)(3 * ROWS_PER_MOD + bn)) * H_DIM + ch;
  union { u16 s[8]; uint4 v; } o;
#pragma unroll
  for (int e = 0; e < 8; ++e) {
    float v = bf2f(bp[e]) + bias[ch + e] + w0 * bf2f(l0[e]) + w1 * bf2f(l1[e]) +
              w2 * bf2f(l2[e]) + w3 * bf2f(l3[e]);
    o.s[e] = f2bf(gelu_exact(v));
  }
  *(uint4*)(x1 + (size_t)r * H_DIM + ch) = o.v;
}

__global__ void combine2_kernel(const float* __restrict__ base,
                                const float* __restrict__ lora,
                                const float* __restrict__ wn,
                                const float* __restrict__ bias,
                                float* __restrict__ out) {
  size_t t = (size_t)blockIdx.x * blockDim.x + threadIdx.x;
  const int chunks = C_DIM / 4;  // 192
  if (t >= (size_t)ROWS * chunks) return;
  int r = (int)(t / chunks);
  int ch = (int)(t % chunks) * 4;
  int bn = r & (ROWS_PER_MOD - 1);
  float w[4] = { wn[r * 4 + 0], wn[r * 4 + 1], wn[r * 4 + 2], wn[r * 4 + 3] };
  float4 bv = *(const float4*)(base + (size_t)r * C_DIM + ch);
  float4 bb = *(const float4*)(bias + ch);
  float4 acc = { bv.x + bb.x, bv.y + bb.y, bv.z + bb.z, bv.w + bb.w };
#pragma unroll
  for (int k = 0; k < 4; ++k) {
    float4 lv = *(const float4*)(lora + ((size_t)(k * ROWS_PER_MOD + bn)) * C_DIM + ch);
    acc.x += w[k] * lv.x; acc.y += w[k] * lv.y;
    acc.z += w[k] * lv.z; acc.w += w[k] * lv.w;
  }
  *(float4*)(out + (size_t)r * C_DIM + ch) = acc;
}

// ---------------------------------------------------------------- launch

extern "C" void kernel_launch(void* const* d_in, const int* in_sizes, int n_in,
                              void* d_out, int out_size, void* d_ws, size_t ws_size,
                              hipStream_t stream) {
  const float* x      = (const float*)d_in[0];
  const float* fc1_w  = (const float*)d_in[1];
  const float* fc1_b  = (const float*)d_in[2];
  const float* fc2_w  = (const float*)d_in[3];
  const float* fc2_b  = (const float*)d_in[4];
  const float* a_fc1  = (const float*)d_in[5];
  const float* b_fc1  = (const float*)d_in[6];
  const float* a_fc2  = (const float*)d_in[7];
  const float* b_fc2  = (const float*)d_in[8];
  const float* gate_w = (const float*)d_in[9];
  const float* gate_b = (const float*)d_in[10];
  const int*   mmask  = (const int*)d_in[11];
  float* out = (float*)d_out;

  char* ws = (char*)d_ws;
  const size_t off_xb     = 0;                         // 16384x768 bf16
  const size_t off_w1b    = off_xb     + (size_t)ROWS * C_DIM * 2;
  const size_t off_d1b    = off_w1b    + (size_t)H_DIM * C_DIM * 2;
  const size_t off_w2b    = off_d1b    + (size_t)M_MOD * H_DIM * C_DIM * 2;
  const size_t off_d2b    = off_w2b    + (size_t)C_DIM * H_DIM * 2;
  const size_t off_logit  = off_d2b    + (size_t)M_MOD * C_DIM * H_DIM * 2;
  const size_t off_wnorm  = off_logit  + (size_t)ROWS * 4 * 4;
  const size_t off_base1  = off_wnorm  + (size_t)ROWS * 4 * 4;
  const size_t off_lora1  = off_base1  + (size_t)ROWS * H_DIM * 2;
  const size_t off_x1     = off_lora1  + (size_t)ROWS * H_DIM * 2;
  const size_t off_base2  = off_base1;                 // reuse (fp32 fits in bf16 region)
  const size_t off_lora2  = off_lora1;

  u16*   xb    = (u16*)(ws + off_xb);
  u16*   w1b   = (u16*)(ws + off_w1b);
  u16*   d1b   = (u16*)(ws + off_d1b);
  u16*   w2b   = (u16*)(ws + off_w2b);
  u16*   d2b   = (u16*)(ws + off_d2b);
  float* logit = (float*)(ws + off_logit);
  float* wnorm = (float*)(ws + off_wnorm);
  u16*   base1 = (u16*)(ws + off_base1);
  u16*   lora1 = (u16*)(ws + off_lora1);
  u16*   x1b   = (u16*)(ws + off_x1);
  float* base2 = (float*)(ws + off_base2);
  float* lora2 = (float*)(ws + off_lora2);

  // 1) casts
  {
    int n4 = ROWS * C_DIM / 4;
    cast_f32_to_bf16_kernel<<<(n4 + 255) / 256, 256, 0, stream>>>(x, xb, n4);
    n4 = H_DIM * C_DIM / 4;
    cast_f32_to_bf16_kernel<<<(n4 + 255) / 256, 256, 0, stream>>>(fc1_w, w1b, n4);
    cast_f32_to_bf16_kernel<<<(n4 + 255) / 256, 256, 0, stream>>>(fc2_w, w2b, n4);
  }
  // 2) HiRA deltas (W ⊙ B@A), bf16
  {
    size_t tot = (size_t)M_MOD * H_DIM * C_DIM;
    make_delta_kernel<<<(unsigned)((tot + 255) / 256), 256, 0, stream>>>(
        fc1_w, a_fc1, b_fc1, d1b, H_DIM, C_DIM);
    make_delta_kernel<<<(unsigned)((tot + 255) / 256), 256, 0, stream>>>(
        fc2_w, a_fc2, b_fc2, d2b, C_DIM, H_DIM);
  }
  // 3) gating
  gate_logits_kernel<<<(ROWS * 4 + 255) / 256, 256, 0, stream>>>(x, gate_w, gate_b, logit);
  gate_norm_kernel<<<(ROWS + 255) / 256, 256, 0, stream>>>(logit, mmask, wnorm);
  // 4) GEMM1 (base + delta share A tile)
  gemm_dual_kernel<true><<<dim3(ROWS / 128, H_DIM / 128), 256, 0, stream>>>(
      xb, w1b, d1b, (void*)base1, (void*)lora1, C_DIM, H_DIM);
  // 5) combine + GELU -> x1 (bf16)
  {
    size_t tot = (size_t)ROWS * (H_DIM / 8);
    combine1_kernel<<<(unsigned)((tot + 255) / 256), 256, 0, stream>>>(
        base1, lora1, wnorm, fc1_b, x1b);
  }
  // 6) GEMM2
  gemm_dual_kernel<false><<<dim3(ROWS / 128, C_DIM / 128), 256, 0, stream>>>(
      x1b, w2b, d2b, (void*)base2, (void*)lora2, H_DIM, C_DIM);
  // 7) combine -> output (fp32)
  {
    size_t tot = (size_t)ROWS * (C_DIM / 4);
    combine2_kernel<<<(unsigned)((tot + 255) / 256), 256, 0, stream>>>(
        base2, lora2, wnorm, fc2_b, out);
  }
}
